// ner_unit_54022098649171
// MI455X (gfx1250) — matrix operand order repaired and checked by reading, not compiled, not run on hardware
//
#include <hip/hip_runtime.h>
#include <hip/hip_bf16.h>
#include <math.h>

// ---- shapes (fixed by reference) ----
#define LL 128
#define BB 8
#define HH 768
#define TT 8

typedef __attribute__((ext_vector_type(16))) __bf16 v16bf;
typedef __attribute__((ext_vector_type(8)))  __bf16 v8bf;
typedef __attribute__((ext_vector_type(8)))  float  v8f;

union BF16x16 { v16bf v; v8bf h[2]; };

__device__ __forceinline__ float elu_fast(float x) {
    return x > 0.f ? x : __expf(x) - 1.f;
}

// ---------------------------------------------------------------------------
// elementwise f32 -> bf16 convert
// ---------------------------------------------------------------------------
__global__ void cvt_bf16(const float* __restrict__ in, __bf16* __restrict__ out, int n) {
    int i = blockIdx.x * blockDim.x + threadIdx.x;
    if (i < n) out[i] = (__bf16)in[i];
}

// build concat([h1,h2], axis=-1) as bf16, rows = L*B, row length 2H
__global__ void concat_cvt(const float* __restrict__ a, const float* __restrict__ b,
                           __bf16* __restrict__ out, int rows) {
    int i = blockIdx.x * blockDim.x + threadIdx.x;
    int total = rows * (2 * HH);
    if (i >= total) return;
    int r = i / (2 * HH), k = i - r * (2 * HH);
    float v = (k < HH) ? a[(size_t)r * HH + k] : b[(size_t)r * HH + (k - HH)];
    out[i] = (__bf16)v;
}

// tag_w (8 x 768) -> bf16 padded to (16 x 768), rows 8..15 zero
__global__ void tagpad_k(const float* __restrict__ tw, __bf16* __restrict__ twp) {
    int i = blockIdx.x * blockDim.x + threadIdx.x;
    if (i >= 16 * HH) return;
    int r = i / HH;
    twp[i] = (__bf16)(r < TT ? tw[i] : 0.f);
}

// ---------------------------------------------------------------------------
// bf16 WMMA GEMM:
//   C[m,n] = act( sum_k A[m,k]*W[n,wofs+k] + bias[n] + gbias[(m&7)*HH + n] )
// one wave per 16x16 C tile, K steps of 32 (v_wmma_f32_16x16x32_bf16).
// ---------------------------------------------------------------------------
__global__ __launch_bounds__(256) void gemm_bf16(
    const __bf16* __restrict__ A, int lda,
    const __bf16* __restrict__ W, int ldw, int wofs,
    float* __restrict__ C, int ldc,
    int K, int Mtiles, int Ntiles,
    const float* __restrict__ bias, const float* __restrict__ gbias, int act_tanh)
{
    int wave = threadIdx.x >> 5;
    int lane = threadIdx.x & 31;
    int tile = blockIdx.x * 8 + wave;
    if (tile >= Mtiles * Ntiles) return;           // wave-uniform guard
    int tn = tile / Mtiles;
    int tm = tile - tn * Mtiles;

    int half = lane >> 4;
    int l16  = lane & 15;
    int row  = tm * 16 + l16;
    int col  = tn * 16 + l16;

    const __bf16* Arow = A + (size_t)row * lda;
    const __bf16* Wrow = W + (size_t)col * ldw + wofs;

    v8f acc = {};
    for (int k0 = 0; k0 < K; k0 += 32) {
        BF16x16 a, b;
        a.h[0] = *(const v8bf*)(Arow + k0 + half * 8);
        a.h[1] = *(const v8bf*)(Arow + k0 + 16 + half * 8);
        b.h[0] = *(const v8bf*)(Wrow + k0 + half * 8);
        b.h[1] = *(const v8bf*)(Wrow + k0 + 16 + half * 8);
        acc = __builtin_amdgcn_wmma_f32_16x16x32_bf16(
            false, a.v, false, b.v, (short)0, acc, false, false);
    }

    float bn = bias ? bias[col] : 0.0f;
    #pragma unroll
    for (int v = 0; v < 8; ++v) {
        int m = tm * 16 + v + half * 8;            // C layout: M = v + 8*laneGroup
        float c = acc[v] + bn;
        if (gbias) c += gbias[(size_t)(m & 7) * HH + col];
        if (act_tanh) c = tanhf(c);
        C[(size_t)m * ldc + col] = c;
    }
}

// ---------------------------------------------------------------------------
// g[b,h] = max over l of hg[(l*B+b), h]
// ---------------------------------------------------------------------------
__global__ void gmax_k(const float* __restrict__ hg, float* __restrict__ g) {
    int i = blockIdx.x * blockDim.x + threadIdx.x;
    if (i >= BB * HH) return;
    int b = i / HH, h = i - b * HH;
    float m = -3.4e38f;
    for (int l = 0; l < LL; ++l)
        m = fmaxf(m, hg[((size_t)(l * BB + b)) * HH + h]);
    g[i] = m;
}

// ---------------------------------------------------------------------------
// a_g[b,k] = sum_h g[b,h] * hid2hid_w[k, 2H + h] + hid2hid_b[k]   (f32)
// ---------------------------------------------------------------------------
__global__ void ag_k(const float* __restrict__ g, const float* __restrict__ hw,
                     const float* __restrict__ hb, float* __restrict__ ag) {
    int i = blockIdx.x * blockDim.x + threadIdx.x;
    if (i >= BB * HH) return;
    int b = i / HH, k = i - b * HH;
    const float* gr = g + (size_t)b * HH;
    const float* wr = hw + (size_t)k * (3 * HH) + 2 * HH;
    float s = 0.0f;
    for (int h = 0; h < HH; ++h) s = fmaf(gr[h], wr[h], s);
    ag[i] = s + hb[k];
}

// ---------------------------------------------------------------------------
// per-row stats: ostats[row] = ( sum_h a[row,h], sum_h a[row,h]^2 )
// one wave per row; 1024 rows.
// ---------------------------------------------------------------------------
__global__ __launch_bounds__(256) void rowstats_k(const float* __restrict__ a,
                                                  float2* __restrict__ ostats) {
    int wave = threadIdx.x >> 5, lane = threadIdx.x & 31;
    int row = blockIdx.x * 8 + wave;
    const float4* pr = (const float4*)(a + (size_t)row * HH + lane * 24);
    float s = 0.f, q = 0.f;
    #pragma unroll
    for (int c = 0; c < 6; ++c) {
        float4 v = pr[c];
        s += v.x + v.y + v.z + v.w;
        q += v.x * v.x + v.y * v.y + v.z * v.z + v.w * v.w;
    }
    #pragma unroll
    for (int off = 16; off; off >>= 1) {
        s += __shfl_xor(s, off, 32);
        q += __shfl_xor(q, off, 32);
    }
    if (lane == 0) ostats[row] = make_float2(s, q);
}

// ---------------------------------------------------------------------------
// cross[b,i,j] = sum_h astg[(i*B+b),h] * aen[(j*B+b),h]
// ---------------------------------------------------------------------------
__global__ void cross_k(const float* __restrict__ astg, const float* __restrict__ aen,
                        float* __restrict__ cross) {
    int idx = blockIdx.x * blockDim.x + threadIdx.x;   // ((b*L+i)*L+j)
    if (idx >= BB * LL * LL) return;
    int j = idx & (LL - 1);
    int r = idx >> 7;
    int i = r & (LL - 1);
    int b = r >> 7;
    const float4* ps = (const float4*)(astg + ((size_t)i * BB + b) * HH);
    const float4* pe = (const float4*)(aen  + ((size_t)j * BB + b) * HH);
    float s = 0.f;
    #pragma unroll 4
    for (int c = 0; c < HH / 4; ++c) {
        float4 x = ps[c], y = pe[c];
        s += x.x * y.x + x.y * y.y + x.z * y.z + x.w * y.w;
    }
    cross[idx] = s;
}

// ---------------------------------------------------------------------------
// Fused tail with WMMA tag projection.
// One wave per (i, b, 16-j tile):  rows j0..j0+15.
// LN stats precomputed:  mu = (Ss+Se)/H ; E2 = (Qs+Qe+2*cross)/H.
// A-fragment (16x32 bf16) built directly in register layout:
//   lane m / m+16 both serve row m; lane holds h = k0 + hi*8 + {0..7, 16..23}.
// B-fragment = padded bf16 tag_w^T (cols t, rows h).
// ---------------------------------------------------------------------------
__global__ __launch_bounds__(256) void fused_wmma(
    const float* __restrict__ astg, const float* __restrict__ aen,
    const float2* __restrict__ st_stats, const float2* __restrict__ en_stats,
    const float* __restrict__ cross,
    const float* __restrict__ lng, const float* __restrict__ lnb,
    const __bf16* __restrict__ twp, const float* __restrict__ tb,
    const float* __restrict__ mask, float* __restrict__ out)
{
    int wave = threadIdx.x >> 5, lane = threadIdx.x & 31;
    int tile = blockIdx.x * 8 + wave;      // ((i*8 + jt)*8 + b), 8192 tiles
    int b  = tile & 7;
    int r1 = tile >> 3;
    int j0 = (r1 & 7) * 16;
    int i  = r1 >> 3;

    if (j0 + 15 < i) {                     // whole tile below diagonal -> zeros
        int r = lane >> 1;
        float4 z = make_float4(0.f, 0.f, 0.f, 0.f);
        *(float4*)(out + (((size_t)(i * LL + j0 + r)) * BB + b) * TT + (lane & 1) * 4) = z;
        return;
    }

    int l16 = lane & 15, hi = lane >> 4;
    int j = j0 + l16;                      // this lane's A row

    // per-row LN statistics (precomputed, no reductions here)
    float2 ss = st_stats[i * BB + b];
    float2 es = en_stats[j * BB + b];
    float  cr = cross[(((size_t)b * LL + i) * LL) + j];
    const float inv = 1.0f / (float)HH;
    float mu   = (ss.x + es.x) * inv;
    float e2   = (ss.y + es.y + 2.f * cr) * inv;
    float rstd = rsqrtf(e2 - mu * mu + 1e-5f);

    const float* srow = astg + ((size_t)i * BB + b) * HH;   // same for all lanes
    const float* erow = aen  + ((size_t)j * BB + b) * HH;

    const __bf16* trow = twp + (size_t)l16 * HH;            // col t = l16 (pad->0)

    v8f acc = {};
    for (int k0 = 0; k0 < HH; k0 += 32) {
        BF16x16 a, bt;
        #pragma unroll
        for (int c = 0; c < 2; ++c) {                       // two 8-wide h chunks
            int hb = k0 + hi * 8 + c * 16;
            float4 s0 = *(const float4*)(srow + hb);
            float4 s1 = *(const float4*)(srow + hb + 4);
            float4 e0 = *(const float4*)(erow + hb);
            float4 e1 = *(const float4*)(erow + hb + 4);
            float4 g0 = *(const float4*)(lng + hb);
            float4 g1 = *(const float4*)(lng + hb + 4);
            float4 d0 = *(const float4*)(lnb + hb);
            float4 d1 = *(const float4*)(lnb + hb + 4);
            float x0 = elu_fast(fmaf((s0.x + e0.x - mu) * rstd, g0.x, d0.x));
            float x1 = elu_fast(fmaf((s0.y + e0.y - mu) * rstd, g0.y, d0.y));
            float x2 = elu_fast(fmaf((s0.z + e0.z - mu) * rstd, g0.z, d0.z));
            float x3 = elu_fast(fmaf((s0.w + e0.w - mu) * rstd, g0.w, d0.w));
            float x4 = elu_fast(fmaf((s1.x + e1.x - mu) * rstd, g1.x, d1.x));
            float x5 = elu_fast(fmaf((s1.y + e1.y - mu) * rstd, g1.y, d1.y));
            float x6 = elu_fast(fmaf((s1.z + e1.z - mu) * rstd, g1.z, d1.z));
            float x7 = elu_fast(fmaf((s1.w + e1.w - mu) * rstd, g1.w, d1.w));
            a.v[c * 8 + 0] = (__bf16)x0;  a.v[c * 8 + 1] = (__bf16)x1;
            a.v[c * 8 + 2] = (__bf16)x2;  a.v[c * 8 + 3] = (__bf16)x3;
            a.v[c * 8 + 4] = (__bf16)x4;  a.v[c * 8 + 5] = (__bf16)x5;
            a.v[c * 8 + 6] = (__bf16)x6;  a.v[c * 8 + 7] = (__bf16)x7;
        }
        bt.h[0] = *(const v8bf*)(trow + k0 + hi * 8);
        bt.h[1] = *(const v8bf*)(trow + k0 + 16 + hi * 8);
        acc = __builtin_amdgcn_wmma_f32_16x16x32_bf16(
            false, a.v, false, bt.v, (short)0, acc, false, false);
    }

    if (l16 < TT) {                        // valid tag columns only
        int t = l16;
        float mi  = mask[i * BB + b];
        float tbt = tb[t];
        #pragma unroll
        for (int v = 0; v < 8; ++v) {
            int jr = j0 + v + hi * 8;      // C layout: M = v + 8*laneGroup
            float m = (jr >= i) ? mi * mask[jr * BB + b] : 0.f;
            float z = acc[v] + tbt;
            out[(((size_t)(i * LL + jr)) * BB + b) * TT + t] = m / (1.f + __expf(-z));
        }
    }
}

// ---------------------------------------------------------------------------
extern "C" void kernel_launch(void* const* d_in, const int* in_sizes, int n_in,
                              void* d_out, int out_size, void* d_ws, size_t ws_size,
                              hipStream_t stream) {
    const float* h1   = (const float*)d_in[0];
    const float* h2   = (const float*)d_in[1];
    const float* mask = (const float*)d_in[2];
    const float* n_w  = (const float*)d_in[3];
    const float* n_b  = (const float*)d_in[4];
    const float* hw   = (const float*)d_in[5];
    const float* hb   = (const float*)d_in[6];
    const float* lng  = (const float*)d_in[7];
    const float* lnb  = (const float*)d_in[8];
    const float* tw   = (const float*)d_in[9];
    const float* tb   = (const float*)d_in[10];
    float* out = (float*)d_out;

    const int MR = LL * BB;                       // 1024 GEMM rows
    char* w = (char*)d_ws;
    size_t o = 0;
    auto take = [&](size_t bytes) { void* p = w + o; o = (o + bytes + 255) & ~(size_t)255; return p; };

    __bf16* cA   = (__bf16*)take((size_t)MR * 2 * HH * 2);
    __bf16* h2b  = (__bf16*)take((size_t)MR * HH * 2);
    __bf16* wnb  = (__bf16*)take((size_t)HH * 2 * HH * 2);
    __bf16* hwb  = (__bf16*)take((size_t)HH * 3 * HH * 2);
    float*  hg   = (float*)take((size_t)MR * HH * 4);
    float*  g    = (float*)take((size_t)BB * HH * 4);
    float*  ag   = (float*)take((size_t)BB * HH * 4);
    float*  astg = (float*)take((size_t)MR * HH * 4);      // a_st + a_g[b] folded
    float*  aen  = (float*)take((size_t)MR * HH * 4);
    float2* sst  = (float2*)take((size_t)MR * 8);
    float2* est  = (float2*)take((size_t)MR * 8);
    float*  crs  = (float*)take((size_t)BB * LL * LL * 4);
    __bf16* twp  = (__bf16*)take((size_t)16 * HH * 2);
    (void)ws_size; (void)in_sizes; (void)n_in; (void)out_size;

    const int TPB = 256;
    // --- precision staging ---
    concat_cvt<<<(MR * 2 * HH + TPB - 1) / TPB, TPB, 0, stream>>>(h1, h2, cA, MR);
    cvt_bf16<<<(MR * HH + TPB - 1) / TPB, TPB, 0, stream>>>(h2, h2b, MR * HH);
    cvt_bf16<<<(HH * 2 * HH + TPB - 1) / TPB, TPB, 0, stream>>>(n_w, wnb, HH * 2 * HH);
    cvt_bf16<<<(HH * 3 * HH + TPB - 1) / TPB, TPB, 0, stream>>>(hw, hwb, HH * 3 * HH);
    tagpad_k<<<(16 * HH + TPB - 1) / TPB, TPB, 0, stream>>>(tw, twp);

    const int Mt = MR / 16, Nt = HH / 16;
    const int gemmBlocks = (Mt * Nt + 7) / 8;
    // hg = tanh(concat @ n_w^T + n_b)
    gemm_bf16<<<gemmBlocks, TPB, 0, stream>>>(cA, 2 * HH, wnb, 2 * HH, 0,
                                              hg, HH, 2 * HH, Mt, Nt, n_b, nullptr, 1);
    gmax_k<<<(BB * HH + TPB - 1) / TPB, TPB, 0, stream>>>(hg, g);
    ag_k<<<(BB * HH + TPB - 1) / TPB, TPB, 0, stream>>>(g, hw, hb, ag);
    // astg = h2 @ W1^T + a_g[b]  (gbias folds a_g) ; aen = h2 @ W2^T
    gemm_bf16<<<gemmBlocks, TPB, 0, stream>>>(h2b, HH, hwb, 3 * HH, 0,
                                              astg, HH, HH, Mt, Nt, nullptr, ag, 0);
    gemm_bf16<<<gemmBlocks, TPB, 0, stream>>>(h2b, HH, hwb, 3 * HH, HH,
                                              aen, HH, HH, Mt, Nt, nullptr, nullptr, 0);
    // separable LN statistics
    rowstats_k<<<MR / 8, TPB, 0, stream>>>(astg, sst);
    rowstats_k<<<MR / 8, TPB, 0, stream>>>(aen, est);
    cross_k<<<(BB * LL * LL + TPB - 1) / TPB, TPB, 0, stream>>>(astg, aen, crs);
    // fused LN + ELU + WMMA tag projection + sigmoid + mask
    fused_wmma<<<(LL * 8 * BB) / 8, TPB, 0, stream>>>(astg, aen, sst, est, crs,
                                                      lng, lnb, twp, tb, mask, out);
}
